// Per_Region_Encoding_58918361367241
// MI455X (gfx1250) — matrix operand order repaired and checked by reading, not compiled, not run on hardware
//
#include <hip/hip_runtime.h>
#include <hip/hip_bf16.h>
#include <math.h>

typedef __attribute__((ext_vector_type(16))) _Float16 v16h;
typedef __attribute__((ext_vector_type(8)))  float    v8f;

#define NB 8
#define NC 256
#define NS 8

// ---------------------------------------------------------------------------
// Weight prep: rearrange OIHW f32 conv weights into WMMA A-matrix fragment
// layout (f16). Layout per conv: [co_tile][tap][chunk][lane][h], h=0..15.
// Contiguous K order (16-bit A 16x32): lane<16 holds M=lane, K = h;
// lane>=16 holds M=lane-16 with +16 K offset.  ci = chunk*32 + koff.
// ---------------------------------------------------------------------------
__global__ void k_prep_w(const float* __restrict__ w, _Float16* __restrict__ pw,
                         int taps, int total) {
  int i = blockIdx.x * blockDim.x + threadIdx.x;
  if (i >= total) return;
  int h     = i & 15;
  int lane  = (i >> 4) & 31;
  int chunk = (i >> 9) & 7;
  int rest  = i >> 12;
  int tap   = rest % taps;
  int cot   = rest / taps;
  int koff  = h + ((lane >= 16) ? 16 : 0);
  int ci    = chunk * 32 + koff;
  int co    = cot * 16 + (lane & 15);
  int widx  = (taps == 9) ? (((co * NC + ci) * 3 + tap / 3) * 3 + (tap % 3))
                          : (co * NC + ci);
  pw[i] = (_Float16)w[widx];
}

// NCHW f32 -> NHWC f16
__global__ void k_convert_f16(const float* __restrict__ in, _Float16* __restrict__ out,
                              int H, int W, int total) {
  int i = blockIdx.x * blockDim.x + threadIdx.x;
  if (i >= total) return;
  int HW = H * W;
  int p  = i % HW;
  int c  = (i / HW) % NC;
  int b  = i / (HW * NC);
  int y = p / W, x = p % W;
  out[(((size_t)b * H + y) * W + x) * NC + c] = (_Float16)in[i];
}

// InstanceNorm (biased var) + LeakyReLU(0.1); reads NCHW f32, writes NHWC f16
__global__ void k_inorm_lrelu(const float* __restrict__ x, const float* __restrict__ g,
                              const float* __restrict__ bta, _Float16* __restrict__ out,
                              int H, int W) {
  int bc = blockIdx.x;
  int b = bc / NC, c = bc % NC;
  int HW = H * W;
  const float* xp = x + (size_t)bc * HW;
  __shared__ float s1[256], s2[256];
  float s = 0.f, q = 0.f;
  for (int p = threadIdx.x; p < HW; p += 256) { float v = xp[p]; s += v; q += v * v; }
  s1[threadIdx.x] = s; s2[threadIdx.x] = q; __syncthreads();
  for (int off = 128; off > 0; off >>= 1) {
    if (threadIdx.x < off) { s1[threadIdx.x] += s1[threadIdx.x + off];
                             s2[threadIdx.x] += s2[threadIdx.x + off]; }
    __syncthreads();
  }
  float mean = s1[0] / HW;
  float var  = s2[0] / HW - mean * mean;
  float inv  = rsqrtf(var + 1e-5f);
  float gg = g[c], bb = bta[c];
  for (int p = threadIdx.x; p < HW; p += 256) {
    float v = (xp[p] - mean) * inv * gg + bb;
    v = (v >= 0.f) ? v : 0.1f * v;
    int y = p / W, xx = p % W;
    out[(((size_t)b * H + y) * W + xx) * NC + c] = (_Float16)v;
  }
}

// ---------------------------------------------------------------------------
// Implicit-GEMM conv via v_wmma_f32_16x16x32_f16 with LDS-staged weights.
// Block = one (batch, co_tile, output row): Wout/16 waves, each wave one
// 16co x 16px tile.  The co_tile's full A-fragment set (T*8 KB) is copied
// into LDS once per block with global_load_async_to_lds_b128 (ASYNCcnt),
// then all waves read A via ds_load while streaming B from global NHWC f16.
//  MODE 0: 3x3 conv pad 1 stride 1
//  MODE 1: ConvTranspose k3 s2 p1 op1 (conv over 2x-dilated input, pad(1,2))
//  MODE 2: 1x1 conv + tanh
// accum=1 -> out += (shortcut add)
// ---------------------------------------------------------------------------
template <int MODE>
__global__ void k_conv_wmma(const _Float16* __restrict__ act, const _Float16* __restrict__ pw,
                            const float* __restrict__ bias, float* __restrict__ out,
                            int Hin, int Win, int Hout, int Wout, int accum) {
  constexpr int T = (MODE == 2) ? 1 : 9;
  extern __shared__ _Float16 sA[];          // T * 8 * 512 halves = T*8192 bytes

  int blocksPerB = 16 * Hout;               // co_tiles * rows
  int b   = blockIdx.x / blocksPerB;
  int r   = blockIdx.x % blocksPerB;
  int cot = r / Hout;
  int oy  = r % Hout;

  // ---- async-copy this co_tile's A fragments into LDS ----
  {
    const int bytes = T * 8192;
    unsigned long long gbase =
        (unsigned long long)(pw + (size_t)cot * T * 4096);
    unsigned lbase = (unsigned)(size_t)(&sA[0]);
    int nth = blockDim.x;
    for (int ofs = threadIdx.x * 16; ofs < bytes; ofs += nth * 16) {
      unsigned lds_addr = lbase + (unsigned)ofs;
      unsigned long long ga = gbase + (unsigned long long)ofs;
      asm volatile("global_load_async_to_lds_b128 %0, %1, off"
                   :: "v"(lds_addr), "v"(ga) : "memory");
    }
    asm volatile("s_wait_asynccnt 0" ::: "memory");
    __syncthreads();
  }

  int lane   = threadIdx.x & 31;
  int wave   = threadIdx.x >> 5;            // x-tile index within the row
  int n      = lane & 15;
  int hiHalf = lane >> 4;                   // 0: K 0..15 / M 0..7 ; 1: K 16..31 / M 8..15
  int kb     = hiHalf * 16;
  int ox     = wave * 16 + n;

  int cobase = cot * 16 + 8 * hiHalf;
  v8f acc;
#pragma unroll
  for (int g = 0; g < 8; ++g) acc[g] = bias[cobase + g];

#pragma unroll
  for (int t = 0; t < T; ++t) {
    int sy, sx; bool rowok, ok;
    if (MODE == 0) {
      int ky = t / 3 - 1, kx = t % 3 - 1;
      sy = oy + ky; rowok = (sy >= 0) && (sy < Hin);
      sx = ox + kx; ok    = (sx >= 0) && (sx < Win);
    } else if (MODE == 1) {
      int ky = t / 3, kx = t % 3;
      int dy = oy - 1 + ky;
      rowok = (dy >= 0) && !(dy & 1) && ((dy >> 1) < Hin);
      sy = dy >> 1;
      int dx = ox - 1 + kx;
      ok = (dx >= 0) && !(dx & 1) && ((dx >> 1) < Win);
      sx = dx >> 1;
    } else { sy = oy; sx = ox; rowok = true; ok = true; }
    if (!rowok) continue;

    const _Float16* bp = act + ((((size_t)(b * Hin + sy)) * Win + sx) * NC + kb);
    const _Float16* aLds = sA + ((size_t)t * 8 * 32 + lane) * 16;
#pragma unroll
    for (int ch = 0; ch < 8; ++ch) {
      v16h av = *(const v16h*)(aLds + (size_t)ch * 512);   // ds_load (32B/lane)
      v16h bv = {};
      if (ok) bv = *(const v16h*)(bp + ch * 32);
      acc = __builtin_amdgcn_wmma_f32_16x16x32_f16(
          /*neg_a=*/false, av, /*neg_b=*/false, bv,
          /*c_mod=*/(short)0, acc, /*reuse_a=*/false, /*reuse_b=*/false);
    }
  }

#pragma unroll
  for (int g = 0; g < 8; ++g) {
    float v = acc[g];
    if (MODE == 2) v = tanhf(v);
    int co = cobase + g;
    size_t oi = (((size_t)b * NC + co) * Hout + oy) * Wout + ox;
    if (accum) out[oi] += v; else out[oi] = v;
  }
}

// ---------------------------------------------------------------------------
// Region pooling helpers
// ---------------------------------------------------------------------------
__global__ void k_zero_cnt(int* cnt) {
  if (threadIdx.x < NB * NS) cnt[threadIdx.x] = 0;
}

__global__ void k_smap(const float* __restrict__ seg, int* __restrict__ smap,
                       int* __restrict__ cnt) {
  int i = blockIdx.x * blockDim.x + threadIdx.x;
  if (i >= NB * 128 * 128) return;
  int b = i >> 14;
  int p = i & 16383;
  int yi = p >> 7, xj = p & 127;
  int s = 0;
  for (int k = 0; k < NS; ++k) {
    float v = seg[(((size_t)b * NS + k) * 256 + 2 * yi) * 256 + 2 * xj];
    if (v != 0.f) { s = k; break; }
  }
  smap[i] = s;
  atomicAdd(&cnt[b * NS + s], 1);
}

__global__ void k_region(const float* __restrict__ codes, const int* __restrict__ smap,
                         const int* __restrict__ cnt, float* __restrict__ outcv) {
  int bc = blockIdx.x;
  int b = bc / NC, c = bc % NC;
  const float* cp = codes + (size_t)bc * 16384;
  const int*   sp = smap + (size_t)b * 16384;
  __shared__ float ls[9];
  if (threadIdx.x < 9) ls[threadIdx.x] = 0.f;
  __syncthreads();
  float rs[8] = {0, 0, 0, 0, 0, 0, 0, 0};
  float gs = 0.f;
  for (int p = threadIdx.x; p < 16384; p += 256) {
    float v = cp[p];
    rs[sp[p]] += v;
    gs += v;
  }
  for (int s = 0; s < 8; ++s) if (rs[s] != 0.f) atomicAdd(&ls[s], rs[s]);
  atomicAdd(&ls[8], gs);
  __syncthreads();
  if (threadIdx.x < 9) {
    int s = threadIdx.x;
    float sum = ls[s];
    float res;
    if (s < 8) {
      int cn = cnt[b * NS + s];
      res = sum / fmaxf((float)cn, 1.f);
    } else {
      res = sum * (1.f / 16384.f);
    }
    outcv[((size_t)b * 9 + s) * NC + c] = res;
  }
}

__global__ void k_exist(const int* __restrict__ cnt, float* __restrict__ ex) {
  int i = threadIdx.x;
  if (i < NB * NS) ex[i] = (cnt[i] > 0) ? 1.f : 0.f;
}

// ---------------------------------------------------------------------------
extern "C" void kernel_launch(void* const* d_in, const int* in_sizes, int n_in,
                              void* d_out, int out_size, void* d_ws, size_t ws_size,
                              hipStream_t stream) {
  const float* x0  = (const float*)d_in[0];   // [8,256,32,32]
  const float* seg = (const float*)d_in[1];   // [8,8,256,256]
  const float* in1g0 = (const float*)d_in[2];
  const float* in1b0 = (const float*)d_in[3];
  const float* w1_0  = (const float*)d_in[4];
  const float* b1_0  = (const float*)d_in[5];
  const float* in2g0 = (const float*)d_in[6];
  const float* in2b0 = (const float*)d_in[7];
  const float* w2_0  = (const float*)d_in[8];
  const float* b2_0  = (const float*)d_in[9];
  const float* ws_0  = (const float*)d_in[10];
  const float* bs_0  = (const float*)d_in[11];
  const float* in1g1 = (const float*)d_in[12];
  const float* in1b1 = (const float*)d_in[13];
  const float* w1_1  = (const float*)d_in[14];
  const float* b1_1  = (const float*)d_in[15];
  const float* in2g1 = (const float*)d_in[16];
  const float* in2b1 = (const float*)d_in[17];
  const float* w2_1  = (const float*)d_in[18];
  const float* b2_1  = (const float*)d_in[19];
  const float* ws_1  = (const float*)d_in[20];
  const float* bs_1  = (const float*)d_in[21];
  const float* codew = (const float*)d_in[22];
  const float* codeb = (const float*)d_in[23];

  char* ws = (char*)d_ws;
  _Float16* W16A  = (_Float16*)(ws);                       // 67,108,864 B
  _Float16* W16B  = (_Float16*)(ws + 67108864);            // 67,108,864 B
  float*    F32A  = (float*)   (ws + 134217728);           // 33,554,432 B
  float*    F32B  = (float*)   (ws + 167772160);           // 33,554,432 B
  float*    CODES = (float*)   (ws + 67108864);            // overlaps W16B/F32A/F32B
  _Float16* PW    = (_Float16*)(ws + 201326592);
  _Float16* pw_w1_0 = PW;
  _Float16* pw_w2_0 = PW + 1 * 589824;
  _Float16* pw_ws_0 = PW + 2 * 589824;
  _Float16* pw_w1_1 = PW + 3 * 589824;
  _Float16* pw_w2_1 = PW + 4 * 589824;
  _Float16* pw_ws_1 = PW + 5 * 589824;
  _Float16* pw_code = PW + 6 * 589824;
  int* smap = (int*)(ws + 201326592 + 7208960);
  int* cnt  = (int*)(ws + 201326592 + 7208960 + 524288);

  float* out_cv = (float*)d_out;            // [8,9,256]
  float* out_ex = (float*)d_out + 18432;    // [8,8]
  float* out_x  = (float*)d_out + 18496;    // [8,256,128,128]

  // ---- weight prep ----
  {
    int tot9 = 16 * 9 * 8 * 32 * 16;   // 589,824
    int tot1 = 16 * 1 * 8 * 32 * 16;   // 65,536
    dim3 blk(256);
    hipLaunchKernelGGL(k_prep_w, dim3((tot9 + 255) / 256), blk, 0, stream, w1_0, pw_w1_0, 9, tot9);
    hipLaunchKernelGGL(k_prep_w, dim3((tot9 + 255) / 256), blk, 0, stream, w2_0, pw_w2_0, 9, tot9);
    hipLaunchKernelGGL(k_prep_w, dim3((tot9 + 255) / 256), blk, 0, stream, ws_0, pw_ws_0, 9, tot9);
    hipLaunchKernelGGL(k_prep_w, dim3((tot9 + 255) / 256), blk, 0, stream, w1_1, pw_w1_1, 9, tot9);
    hipLaunchKernelGGL(k_prep_w, dim3((tot9 + 255) / 256), blk, 0, stream, w2_1, pw_w2_1, 9, tot9);
    hipLaunchKernelGGL(k_prep_w, dim3((tot9 + 255) / 256), blk, 0, stream, ws_1, pw_ws_1, 9, tot9);
    hipLaunchKernelGGL(k_prep_w, dim3((tot1 + 255) / 256), blk, 0, stream, codew, pw_code, 1, tot1);
  }
  hipLaunchKernelGGL(k_zero_cnt, dim3(1), dim3(64), 0, stream, cnt);

  const int SH9 = 9 * 8192;   // LDS bytes for 3x3 / convT
  const int SH1 = 1 * 8192;   // LDS bytes for 1x1

  // ---- block 0 (32 -> 64) ----
  {
    int tot = NB * NC * 32 * 32;
    hipLaunchKernelGGL(k_convert_f16, dim3(tot / 256), dim3(256), 0, stream, x0, W16A, 32, 32, tot);
    hipLaunchKernelGGL(k_inorm_lrelu, dim3(NB * NC), dim3(256), 0, stream, x0, in1g0, in1b0, W16B, 32, 32);
    k_conv_wmma<0><<<dim3(NB * 16 * 32), dim3(64), SH9, stream>>>(
        W16B, pw_w1_0, b1_0, F32A, 32, 32, 32, 32, 0);
    hipLaunchKernelGGL(k_inorm_lrelu, dim3(NB * NC), dim3(256), 0, stream, F32A, in2g0, in2b0, W16B, 32, 32);
    k_conv_wmma<1><<<dim3(NB * 16 * 64), dim3(128), SH9, stream>>>(
        W16B, pw_w2_0, b2_0, F32B, 32, 32, 64, 64, 0);
    k_conv_wmma<1><<<dim3(NB * 16 * 64), dim3(128), SH9, stream>>>(
        W16A, pw_ws_0, bs_0, F32B, 32, 32, 64, 64, 1);
  }

  // ---- block 1 (64 -> 128) ----
  {
    int tot = NB * NC * 64 * 64;
    hipLaunchKernelGGL(k_convert_f16, dim3(tot / 256), dim3(256), 0, stream, F32B, W16A, 64, 64, tot);
    hipLaunchKernelGGL(k_inorm_lrelu, dim3(NB * NC), dim3(256), 0, stream, F32B, in1g1, in1b1, W16B, 64, 64);
    k_conv_wmma<0><<<dim3(NB * 16 * 64), dim3(128), SH9, stream>>>(
        W16B, pw_w1_1, b1_1, F32A, 64, 64, 64, 64, 0);
    hipLaunchKernelGGL(k_inorm_lrelu, dim3(NB * NC), dim3(256), 0, stream, F32A, in2g1, in2b1, W16B, 64, 64);
    k_conv_wmma<1><<<dim3(NB * 16 * 128), dim3(256), SH9, stream>>>(
        W16B, pw_w2_1, b2_1, out_x, 64, 64, 128, 128, 0);
    k_conv_wmma<1><<<dim3(NB * 16 * 128), dim3(256), SH9, stream>>>(
        W16A, pw_ws_1, bs_1, out_x, 64, 64, 128, 128, 1);
  }

  // ---- codes: 1x1 conv + tanh ----
  {
    int tot = NB * NC * 128 * 128;
    hipLaunchKernelGGL(k_convert_f16, dim3(tot / 256), dim3(256), 0, stream, out_x, W16A, 128, 128, tot);
    k_conv_wmma<2><<<dim3(NB * 16 * 128), dim3(256), SH1, stream>>>(
        W16A, pw_code, codeb, CODES, 128, 128, 128, 128, 0);
  }

  // ---- region pooling ----
  hipLaunchKernelGGL(k_smap, dim3((NB * 128 * 128) / 256), dim3(256), 0, stream, seg, smap, cnt);
  hipLaunchKernelGGL(k_region, dim3(NB * NC), dim3(256), 0, stream, CODES, smap, cnt, out_cv);
  hipLaunchKernelGGL(k_exist, dim3(1), dim3(64), 0, stream, cnt, out_ex);
}
